// BiDAF_25357486915930
// MI455X (gfx1250) — matrix operand order repaired and checked by reading, not compiled, not run on hardware
//
#include <hip/hip_runtime.h>
#include <hip/hip_bf16.h>
#include <math.h>

// ---------------------------------------------------------------------------
// BiDAF forward for MI455X (gfx1250, wave32, WMMA, bf16 inputs / f32 accum).
// ---------------------------------------------------------------------------

typedef __bf16 bf16x16 __attribute__((ext_vector_type(16)));
typedef float  f32x8   __attribute__((ext_vector_type(8)));

#define H_      256
#define H4      1024
#define B_      32
#define LC_     512
#define LQ_     64
#define WC_     16
#define CEMB    8
#define CHID    100
#define CK      5
#define WEMB    300
#define DIN0P   416   // 100 + 300 padded to mult of 32

__device__ __forceinline__ float sigmoidf_(float x) { return 1.0f / (1.0f + expf(-x)); }

#define WMMA_BF16(a, b, c) \
    __builtin_amdgcn_wmma_f32_16x16x32_bf16(false, (a), false, (b), (short)0, (c), false, false)

// ---- WMMA fragment loaders (wave32; contiguous k order, same for A and B) ----------
// A (16x32 bf16, MxK): lanes 0-15 hold row=lane, K 0..15 contiguous;
// lanes 16-31 hold row=lane-16, K 16..31 contiguous.
__device__ __forceinline__ bf16x16 load_frag_a(const __bf16* base, int ld, int lane) {
    const __bf16* p = base + (size_t)(lane & 15) * ld + ((lane >> 4) << 4);
    bf16x16 v;
#pragma unroll
    for (int j = 0; j < 16; ++j) v[j] = p[j];
    return v;
}
// B (32x16 bf16, KxN) from weight-major W[N][K]: lane n (0-15) holds column n,
// K 0..15 contiguous; lanes 16-31 hold column n-16, K 16..31 contiguous.
__device__ __forceinline__ bf16x16 load_frag_b(const __bf16* base, int ld, int lane) {
    const __bf16* p = base + (size_t)(lane & 15) * ld + ((lane >> 4) << 4);
    bf16x16 v;
#pragma unroll
    for (int j = 0; j < 16; ++j) v[j] = p[j];
    return v;
}
// C/D frag element (r, lane): row = r + 8*(lane>=16), col = lane & 15.

// ---------------------------------------------------------------------------
// Generic batched GEMM: C[z][M][N] = A[z][M][K] * W[z][N][K]^T (+ bias[z][N])
// block = 128 threads (4 waves); block tile 128(M) x 64(N); wave tile 32x64.
// Double-buffered fragments: loads for step s+1 overlap the 8 WMMAs of step s.
// M mult of 128, N mult of 64, K mult of 32.
// ---------------------------------------------------------------------------
__global__ __launch_bounds__(128)
void wmma_gemm_kernel(const __bf16* __restrict__ A, const __bf16* __restrict__ W,
                      const float* __restrict__ bias, float* __restrict__ C,
                      int M, int N, int K,
                      long long sA, long long sW, long long sC, long long sBias) {
    A += (long long)blockIdx.z * sA;
    W += (long long)blockIdx.z * sW;
    C += (long long)blockIdx.z * sC;
    if (bias) bias += (long long)blockIdx.z * sBias;

    const int tid  = threadIdx.x;
    const int lane = tid & 31;
    const int w    = tid >> 5;
    const int m0   = blockIdx.y * 128 + w * 32;
    const int n0   = blockIdx.x * 64;

    const __bf16* Arow0 = A + (size_t)m0 * K;
    const __bf16* Arow1 = A + (size_t)(m0 + 16) * K;
    const __bf16* Wn0 = W + (size_t)(n0 + 0)  * K;
    const __bf16* Wn1 = W + (size_t)(n0 + 16) * K;
    const __bf16* Wn2 = W + (size_t)(n0 + 32) * K;
    const __bf16* Wn3 = W + (size_t)(n0 + 48) * K;

    f32x8 acc[8] = {};
    bf16x16 a0A, a1A, b0A, b1A, b2A, b3A;   // ping
    bf16x16 a0B, a1B, b0B, b1B, b2B, b3B;   // pong

#define LD_SET(SUF, koff)                                   \
    a0##SUF = load_frag_a(Arow0 + (koff), K, lane);         \
    a1##SUF = load_frag_a(Arow1 + (koff), K, lane);         \
    b0##SUF = load_frag_b(Wn0 + (koff), K, lane);           \
    b1##SUF = load_frag_b(Wn1 + (koff), K, lane);           \
    b2##SUF = load_frag_b(Wn2 + (koff), K, lane);           \
    b3##SUF = load_frag_b(Wn3 + (koff), K, lane);
#define MMA_SET(SUF)                                        \
    acc[0] = WMMA_BF16(a0##SUF, b0##SUF, acc[0]);           \
    acc[1] = WMMA_BF16(a0##SUF, b1##SUF, acc[1]);           \
    acc[2] = WMMA_BF16(a0##SUF, b2##SUF, acc[2]);           \
    acc[3] = WMMA_BF16(a0##SUF, b3##SUF, acc[3]);           \
    acc[4] = WMMA_BF16(a1##SUF, b0##SUF, acc[4]);           \
    acc[5] = WMMA_BF16(a1##SUF, b1##SUF, acc[5]);           \
    acc[6] = WMMA_BF16(a1##SUF, b2##SUF, acc[6]);           \
    acc[7] = WMMA_BF16(a1##SUF, b3##SUF, acc[7]);

    LD_SET(A, 0)
    int k0 = 0;
    for (;;) {
        int kn = k0 + 32;
        int kl = (kn < K) ? kn : 0;
        __builtin_prefetch(Wn0 + k0 + 128, 0, 1);
        LD_SET(B, kl)
        MMA_SET(A)
        k0 = kn;
        if (k0 >= K) break;
        kn = k0 + 32;
        kl = (kn < K) ? kn : 0;
        __builtin_prefetch(Arow0 + k0 + 128, 0, 1);
        LD_SET(A, kl)
        MMA_SET(B)
        k0 = kn;
        if (k0 >= K) break;
    }
#undef LD_SET
#undef MMA_SET

    const int coff = lane & 15;
    const int roff = (lane >> 4) << 3;
#pragma unroll
    for (int i = 0; i < 4; ++i) {
        int col = n0 + i * 16 + coff;
        float bb = bias ? bias[col] : 0.0f;
#pragma unroll
        for (int r = 0; r < 8; ++r) {
            C[(size_t)(m0 + r + roff) * N + col]      = acc[i][r] + bb;
            C[(size_t)(m0 + 16 + r + roff) * N + col] = acc[4 + i][r] + bb;
        }
    }
}

// ---------------------------------------------------------------------------
// BiLSTM recurrent scan. grid.x = 2 (dir), block = 1024 (32 waves).
// h (bf16) and c (f32) stay resident in LDS across timesteps. Each wave owns
// m0 in {0,16} (batch half) and a 16-wide h-column group jg, and computes the
// i/f/g/o gate tiles for those columns -> activation is wave-local.
// xg: [dir][B*T, 4H] (bias folded in by the GEMM); added post-GEMM so its
// loads overlap the whole recurrent WMMA chain.
// ---------------------------------------------------------------------------
__global__ __launch_bounds__(1024)
void lstm_scan_kernel(const float* __restrict__ xg, const __bf16* __restrict__ whh,
                      float* __restrict__ out, int T,
                      float* __restrict__ final_hc, int out_ld) {
    __shared__ __bf16 Hst[B_ * H_];   // 16 KB
    __shared__ float  Cst[B_ * H_];   // 32 KB

    const int dir = blockIdx.x;
    xg  += (long long)dir * B_ * T * H4;
    whh += (long long)dir * H4 * H_;
    const int col_off = dir * H_;
    const int fh_off  = dir * H_;          // feat = [hf, hb, cf, cb]
    const int fc_off  = 2 * H_ + dir * H_;

    const int tid  = threadIdx.x;
    const int lane = tid & 31;
    const int w    = tid >> 5;
    const int m0   = (w >> 4) << 4;        // 0 or 16 (batch rows)
    const int jg   = (w & 15) << 4;        // h-column group
    const int coff = lane & 15;
    const int roff = (lane >> 4) << 3;

    const __bf16* Wg0 = whh + (size_t)(0 * H_ + jg) * H_;
    const __bf16* Wg1 = whh + (size_t)(1 * H_ + jg) * H_;
    const __bf16* Wg2 = whh + (size_t)(2 * H_ + jg) * H_;
    const __bf16* Wg3 = whh + (size_t)(3 * H_ + jg) * H_;

    for (int i = tid; i < B_ * H_; i += 1024) { Hst[i] = (__bf16)0.0f; Cst[i] = 0.0f; }
    __syncthreads();

    for (int t = 0; t < T; ++t) {
        const int tt = dir ? (T - 1 - t) : t;
        // issue input-projection loads early; consumed only after the k-loop
        f32x8 xv[4];
#pragma unroll
        for (int g = 0; g < 4; ++g) {
            const int colbase = g * H_ + jg;
#pragma unroll
            for (int r = 0; r < 8; ++r) {
                int brow = m0 + r + roff;
                xv[g][r] = xg[((long long)brow * T + tt) * H4 + colbase + coff];
            }
        }
        f32x8 acc[4] = {};
        bf16x16 b0A, b1A, b2A, b3A, b0B, b1B, b2B, b3B;
#define LDB_SET(SUF, koff)                                  \
        b0##SUF = load_frag_b(Wg0 + (koff), H_, lane);      \
        b1##SUF = load_frag_b(Wg1 + (koff), H_, lane);      \
        b2##SUF = load_frag_b(Wg2 + (koff), H_, lane);      \
        b3##SUF = load_frag_b(Wg3 + (koff), H_, lane);
#define MMA_SCAN(SUF, koff) {                                                  \
        bf16x16 aa = load_frag_a(Hst + m0 * H_ + (koff), H_, lane);            \
        acc[0] = WMMA_BF16(aa, b0##SUF, acc[0]);                               \
        acc[1] = WMMA_BF16(aa, b1##SUF, acc[1]);                               \
        acc[2] = WMMA_BF16(aa, b2##SUF, acc[2]);                               \
        acc[3] = WMMA_BF16(aa, b3##SUF, acc[3]); }

        LDB_SET(A, 0)
#pragma unroll
        for (int s = 0; s < 4; ++s) {
            const int k0 = s * 64;
            LDB_SET(B, k0 + 32)
            MMA_SCAN(A, k0)
            const int kn = (k0 + 64 < H_) ? (k0 + 64) : 0;
            LDB_SET(A, kn)
            MMA_SCAN(B, k0 + 32)
        }
#undef LDB_SET
#undef MMA_SCAN
        __syncthreads();   // all waves done reading Hst before it is rewritten
#pragma unroll
        for (int r = 0; r < 8; ++r) {
            int brow = m0 + r + roff;
            int col  = jg + coff;
            float ig = sigmoidf_(acc[0][r] + xv[0][r]);
            float fg = sigmoidf_(acc[1][r] + xv[1][r]);
            float gg = tanhf(acc[2][r] + xv[2][r]);
            float og = sigmoidf_(acc[3][r] + xv[3][r]);
            float cn = fg * Cst[brow * H_ + col] + ig * gg;
            float hn = og * tanhf(cn);
            Cst[brow * H_ + col] = cn;
            Hst[brow * H_ + col] = (__bf16)hn;
            out[((long long)brow * T + tt) * out_ld + col_off + col] = hn;
            if (final_hc && t == T - 1) {
                final_hc[brow * H4 + fh_off + col] = hn;
                final_hc[brow * H4 + fc_off + col] = cn;
            }
        }
        __syncthreads();
    }
}

// --------------------------- scalar helper kernels -------------------------

__global__ void f32_to_bf16_pad(const float* __restrict__ src, __bf16* __restrict__ dst,
                                int rows, int cols, int cols_p) {
    int n = rows * cols_p;
    for (int i = blockIdx.x * blockDim.x + threadIdx.x; i < n; i += gridDim.x * blockDim.x) {
        int r = i / cols_p, c = i - r * cols_p;
        dst[i] = (c < cols) ? (__bf16)src[(size_t)r * cols + c] : (__bf16)0.0f;
    }
}

// char CNN: embed 16 chars, conv width-5 VALID over 12 positions, maxpool.
__global__ __launch_bounds__(128)
void char_conv_kernel(const int* __restrict__ chars, const float* __restrict__ emb,
                      const float* __restrict__ cw, const float* __restrict__ cb,
                      __bf16* __restrict__ X) {
    __shared__ float E[WC_][CEMB];
    const int bl = blockIdx.x;
    const int tid = threadIdx.x;
    if (tid < WC_ * CEMB) {
        int wpos = tid >> 3, e = tid & 7;
        E[wpos][e] = emb[(size_t)chars[(size_t)bl * WC_ + wpos] * CEMB + e];
    }
    __syncthreads();
    if (tid < CHID) {
        float best = -3.0e38f;
        float bias = cb[tid];
        for (int p = 0; p <= WC_ - CK; ++p) {
            float s = bias;
#pragma unroll
            for (int k = 0; k < CK; ++k)
#pragma unroll
                for (int e = 0; e < CEMB; ++e)
                    s += E[p + k][e] * cw[tid * (CEMB * CK) + e * CK + k];
            best = fmaxf(best, s);
        }
        X[(size_t)bl * DIN0P + tid] = (__bf16)best;
    } else if (tid >= 112 && tid < 128) {
        X[(size_t)bl * DIN0P + 400 + (tid - 112)] = (__bf16)0.0f;  // K pad
    }
}

__global__ void word_gather_kernel(const int* __restrict__ words, const float* __restrict__ wemb,
                                   __bf16* __restrict__ X, int n) {
    for (int i = blockIdx.x * blockDim.x + threadIdx.x; i < n; i += gridDim.x * blockDim.x) {
        int row = i / WEMB, d = i - row * WEMB;
        X[(size_t)row * DIN0P + CHID + d] = (__bf16)wemb[(size_t)words[row] * WEMB + d];
    }
}

__global__ void cwm_kernel(const float* __restrict__ cenc, const float* __restrict__ wm,
                           __bf16* __restrict__ cwm, int n) {
    for (int i = blockIdx.x * blockDim.x + threadIdx.x; i < n; i += gridDim.x * blockDim.x) {
        int h = i & 511;
        cwm[i] = (__bf16)(cenc[i] * wm[h]);
    }
}

__global__ void rowdot_kernel(const float* __restrict__ x, const float* __restrict__ w,
                              float* __restrict__ out, int rows) {
    for (int i = blockIdx.x * blockDim.x + threadIdx.x; i < rows; i += gridDim.x * blockDim.x) {
        float s = 0.0f;
        const float* xr = x + (size_t)i * 512;
        for (int h = 0; h < 512; ++h) s += xr[h] * w[h];
        out[i] = s;
    }
}

// fused: S' = S + rt[row] + ct[col] + alpha_b, softmax over 64 cols -> bf16 A;
// also stores rowmax of S' (input to the q2c softmax).
__global__ __launch_bounds__(64)
void softmax64_kernel(const float* __restrict__ S, const float* __restrict__ rt,
                      const float* __restrict__ ct, const float* __restrict__ alpha_b,
                      __bf16* __restrict__ A, float* __restrict__ rowmax) {
    __shared__ float red[64];
    const int row = blockIdx.x, j = threadIdx.x;
    const int b = row >> 9;   // LC_ == 512
    float v = S[(size_t)row * LQ_ + j] + rt[row] + ct[(size_t)b * LQ_ + j] + alpha_b[0];
    red[j] = v; __syncthreads();
    for (int s = 32; s > 0; s >>= 1) { if (j < s) red[j] = fmaxf(red[j], red[j + s]); __syncthreads(); }
    float m = red[0]; __syncthreads();
    float e = expf(v - m);
    red[j] = e; __syncthreads();
    for (int s = 32; s > 0; s >>= 1) { if (j < s) red[j] += red[j + s]; __syncthreads(); }
    A[(size_t)row * LQ_ + j] = (__bf16)(e / red[0]);
    if (j == 0) rowmax[row] = m;
}

__global__ __launch_bounds__(512)
void softmax512_kernel(const float* __restrict__ rowmax, float* __restrict__ bvec) {
    __shared__ float red[512];
    const int b = blockIdx.x, i = threadIdx.x;
    float v = rowmax[(size_t)b * LC_ + i];
    red[i] = v; __syncthreads();
    for (int s = 256; s > 0; s >>= 1) { if (i < s) red[i] = fmaxf(red[i], red[i + s]); __syncthreads(); }
    float m = red[0]; __syncthreads();
    float e = expf(v - m);
    red[i] = e; __syncthreads();
    for (int s = 256; s > 0; s >>= 1) { if (i < s) red[i] += red[i + s]; __syncthreads(); }
    bvec[(size_t)b * LC_ + i] = e / red[0];
}

__global__ void q2c_kernel(const float* __restrict__ bvec, const float* __restrict__ cenc,
                           float* __restrict__ q2c, int n) {
    for (int i = blockIdx.x * blockDim.x + threadIdx.x; i < n; i += gridDim.x * blockDim.x) {
        int b = i >> 9, h = i & 511;
        float s = 0.0f;
        for (int p = 0; p < LC_; ++p)
            s += bvec[(size_t)b * LC_ + p] * cenc[((size_t)b * LC_ + p) * 512 + h];
        q2c[i] = s;
    }
}

__global__ void qT_kernel(const float* __restrict__ qenc, __bf16* __restrict__ qT, int n) {
    for (int i = blockIdx.x * blockDim.x + threadIdx.x; i < n; i += gridDim.x * blockDim.x) {
        int b = i / (512 * LQ_);
        int rem = i - b * (512 * LQ_);
        int h = rem / LQ_, j = rem - h * LQ_;
        qT[i] = (__bf16)qenc[((size_t)b * LQ_ + j) * 512 + h];
    }
}

__global__ void g_build_kernel(const float* __restrict__ cenc, const float* __restrict__ c2q,
                               const float* __restrict__ q2c, __bf16* __restrict__ g, int n) {
    for (int i = blockIdx.x * blockDim.x + threadIdx.x; i < n; i += gridDim.x * blockDim.x) {
        int row = i >> 9, h = i & 511;
        int b = row >> 9;  // LC_ == 512
        float c = cenc[i], cq = c2q[i];
        __bf16* gr = g + (size_t)row * 2048;
        gr[h]        = (__bf16)c;
        gr[512 + h]  = (__bf16)cq;
        gr[1024 + h] = (__bf16)(c * cq);
        gr[1536 + h] = (__bf16)(c * q2c[(size_t)b * 512 + h]);
    }
}

__global__ void gm_build_kernel(const __bf16* __restrict__ g, const float* __restrict__ m2,
                                __bf16* __restrict__ gm, int n) {
    for (int i = blockIdx.x * blockDim.x + threadIdx.x; i < n; i += gridDim.x * blockDim.x) {
        int row = i / 2560, col = i - row * 2560;
        gm[i] = (col < 2048) ? g[(size_t)row * 2048 + col]
                             : (__bf16)m2[(size_t)row * 512 + (col - 2048)];
    }
}

__global__ __launch_bounds__(256)
void clf_kernel(const float* __restrict__ feat, const float* __restrict__ w,
                const float* __restrict__ cb, float* __restrict__ out) {
    __shared__ float red[256];
    const int b = blockIdx.x, t = threadIdx.x;
    float s = 0.0f;
    for (int k = t; k < H4; k += 256) s += feat[(size_t)b * H4 + k] * w[k];
    red[t] = s; __syncthreads();
    for (int st = 128; st > 0; st >>= 1) { if (t < st) red[t] += red[t + st]; __syncthreads(); }
    if (t == 0) out[b] = red[0] + cb[0];
}

// ---------------------------------------------------------------------------
extern "C" void kernel_launch(void* const* d_in, const int* in_sizes, int n_in,
                              void* d_out, int out_size, void* d_ws, size_t ws_size,
                              hipStream_t stream) {
    (void)in_sizes; (void)n_in; (void)out_size; (void)ws_size;
    const int*   c_word   = (const int*)d_in[0];
    const int*   q_word   = (const int*)d_in[1];
    const int*   c_char   = (const int*)d_in[2];
    const int*   q_char   = (const int*)d_in[3];
    const float* char_emb = (const float*)d_in[4];
    const float* conv_w   = (const float*)d_in[5];
    const float* conv_b   = (const float*)d_in[6];
    const float* word_emb = (const float*)d_in[7];
    const float* ctx_Wih  = (const float*)d_in[8];
    const float* ctx_Whh  = (const float*)d_in[9];
    const float* ctx_b    = (const float*)d_in[10];
    const float* alpha_w  = (const float*)d_in[11];
    const float* alpha_b  = (const float*)d_in[12];
    const float* mod1_Wih = (const float*)d_in[13];
    const float* mod1_Whh = (const float*)d_in[14];
    const float* mod1_b   = (const float*)d_in[15];
    const float* mod2_Wih = (const float*)d_in[16];
    const float* mod2_Whh = (const float*)d_in[17];
    const float* mod2_b   = (const float*)d_in[18];
    const float* out_Wih  = (const float*)d_in[19];
    const float* out_Whh  = (const float*)d_in[20];
    const float* out_b    = (const float*)d_in[21];
    const float* clf_w    = (const float*)d_in[22];
    const float* clf_b    = (const float*)d_in[23];
    float* out = (float*)d_out;

    // ---- workspace carve ----
    size_t off = 0;
    auto alloc = [&](size_t bytes) -> char* {
        char* r = (char*)d_ws + off;
        off = (off + bytes + 255) & ~(size_t)255;
        return r;
    };
    const long long MC = (long long)B_ * LC_;   // 16384
    const long long MQ = (long long)B_ * LQ_;   // 2048
    __bf16* Xc     = (__bf16*)alloc(MC * DIN0P * 2);
    __bf16* Xq     = (__bf16*)alloc(MQ * DIN0P * 2);
    float*  XG     = (float*)alloc(2 * MC * H4 * 4);   // both directions
    float*  cenc   = (float*)alloc(MC * 512 * 4);
    float*  qenc   = (float*)alloc(MQ * 512 * 4);
    __bf16* cwm    = (__bf16*)alloc(MC * 512 * 2);
    __bf16* qbf    = (__bf16*)alloc(MQ * 512 * 2);
    float*  S      = (float*)alloc(MC * LQ_ * 4);
    __bf16* Abf    = (__bf16*)alloc(MC * LQ_ * 2);
    float*  rowmax = (float*)alloc(MC * 4);
    float*  bvec   = (float*)alloc(MC * 4);
    float*  rt     = (float*)alloc(MC * 4);
    float*  ct     = (float*)alloc(MQ * 4);
    __bf16* qT     = (__bf16*)alloc((long long)B_ * 512 * LQ_ * 2);
    float*  c2q    = (float*)alloc(MC * 512 * 4);      // later reused as out-LSTM h dump
    float*  q2c    = (float*)alloc((long long)B_ * 512 * 4);
    __bf16* gbf    = (__bf16*)alloc(MC * 2048 * 2);
    float*  m1     = (float*)alloc(MC * 512 * 4);
    __bf16* m1bf   = (__bf16*)alloc(MC * 512 * 2);
    float*  m2     = (float*)alloc(MC * 512 * 4);
    __bf16* gmbf   = (__bf16*)alloc(MC * 2560 * 2);
    float*  feat   = (float*)alloc((long long)B_ * H4 * 4);
    __bf16* ctxWihB = (__bf16*)alloc(2LL * H4 * DIN0P * 2);
    __bf16* ctxWhhB = (__bf16*)alloc(2LL * H4 * H_ * 2);
    __bf16* m1WihB  = (__bf16*)alloc(2LL * H4 * 2048 * 2);
    __bf16* m1WhhB  = (__bf16*)alloc(2LL * H4 * H_ * 2);
    __bf16* m2WihB  = (__bf16*)alloc(2LL * H4 * 512 * 2);
    __bf16* m2WhhB  = (__bf16*)alloc(2LL * H4 * H_ * 2);
    __bf16* oWihB   = (__bf16*)alloc(2LL * H4 * 2560 * 2);
    __bf16* oWhhB   = (__bf16*)alloc(2LL * H4 * H_ * 2);

    auto gs = [](long long n) { return dim3((unsigned)((n + 255) / 256)); };
    auto cvt = [&](const float* src, __bf16* dst, int rows, int cols, int colsp) {
        f32_to_bf16_pad<<<gs((long long)rows * colsp), 256, 0, stream>>>(src, dst, rows, cols, colsp);
    };

    // ---- weights -> bf16 (pad K to mult of 32) ----
    cvt(ctx_Wih,  ctxWihB, 2 * H4, 400,  DIN0P);
    cvt(ctx_Whh,  ctxWhhB, 2 * H4, 256,  256);
    cvt(mod1_Wih, m1WihB,  2 * H4, 2048, 2048);
    cvt(mod1_Whh, m1WhhB,  2 * H4, 256,  256);
    cvt(mod2_Wih, m2WihB,  2 * H4, 512,  512);
    cvt(mod2_Whh, m2WhhB,  2 * H4, 256,  256);
    cvt(out_Wih,  oWihB,   2 * H4, 2560, 2560);
    cvt(out_Whh,  oWhhB,   2 * H4, 256,  256);

    // ---- char CNN + word embedding -> bf16 inputs ----
    char_conv_kernel<<<dim3((unsigned)MC), 128, 0, stream>>>(c_char, char_emb, conv_w, conv_b, Xc);
    char_conv_kernel<<<dim3((unsigned)MQ), 128, 0, stream>>>(q_char, char_emb, conv_w, conv_b, Xq);
    word_gather_kernel<<<gs(MC * WEMB), 256, 0, stream>>>(c_word, word_emb, Xc, (int)(MC * WEMB));
    word_gather_kernel<<<gs(MQ * WEMB), 256, 0, stream>>>(q_word, word_emb, Xq, (int)(MQ * WEMB));

    // ---- BiLSTM helper: input-proj GEMM (both dirs batched) + scan ----
    auto bilstm = [&](const __bf16* X, int M, int T, const __bf16* WihB, const __bf16* WhhB,
                      const float* bias, int Kp, float* outbuf, int out_ld, float* final_hc) {
        dim3 g(H4 / 64, M / 128, 2);
        wmma_gemm_kernel<<<g, 128, 0, stream>>>(X, WihB, bias, XG, M, H4, Kp,
                                                0LL, (long long)H4 * Kp,
                                                (long long)M * H4, (long long)H4);
        lstm_scan_kernel<<<dim3(2), 1024, 0, stream>>>(XG, WhhB, outbuf, T, final_hc, out_ld);
    };

    // ---- context encoder ----
    bilstm(Xc, (int)MC, LC_, ctxWihB, ctxWhhB, ctx_b, DIN0P, cenc, 512, nullptr);
    bilstm(Xq, (int)MQ, LQ_, ctxWihB, ctxWhhB, ctx_b, DIN0P, qenc, 512, nullptr);

    // ---- attention ----
    cwm_kernel<<<gs(MC * 512), 256, 0, stream>>>(cenc, alpha_w + 1024, cwm, (int)(MC * 512));
    f32_to_bf16_pad<<<gs(MQ * 512), 256, 0, stream>>>(qenc, qbf, (int)MQ, 512, 512);
    rowdot_kernel<<<gs(MC), 256, 0, stream>>>(cenc, alpha_w,       rt, (int)MC);
    rowdot_kernel<<<gs(MQ), 256, 0, stream>>>(qenc, alpha_w + 512, ct, (int)MQ);
    // S[b] = cwm[b] @ q[b]^T : M=512, N=64, K=512
    wmma_gemm_kernel<<<dim3(1, LC_ / 128, B_), 128, 0, stream>>>(
        cwm, qbf, nullptr, S, LC_, LQ_, 512,
        (long long)LC_ * 512, (long long)LQ_ * 512, (long long)LC_ * LQ_, 0LL);
    softmax64_kernel<<<dim3((unsigned)MC), 64, 0, stream>>>(S, rt, ct, alpha_b, Abf, rowmax);
    softmax512_kernel<<<dim3(B_), 512, 0, stream>>>(rowmax, bvec);
    q2c_kernel<<<gs((long long)B_ * 512), 256, 0, stream>>>(bvec, cenc, q2c, B_ * 512);
    qT_kernel<<<gs((long long)B_ * 512 * LQ_), 256, 0, stream>>>(qenc, qT, B_ * 512 * LQ_);
    // c2q = a @ q : M=512, N=512, K=64
    wmma_gemm_kernel<<<dim3(512 / 64, 512 / 128, B_), 128, 0, stream>>>(
        Abf, qT, nullptr, c2q, 512, 512, 64,
        (long long)LC_ * LQ_, (long long)512 * LQ_, (long long)LC_ * 512, 0LL);
    g_build_kernel<<<gs(MC * 512), 256, 0, stream>>>(cenc, c2q, q2c, gbf, (int)(MC * 512));

    // ---- modeling layers ----
    bilstm(gbf, (int)MC, LC_, m1WihB, m1WhhB, mod1_b, 2048, m1, 512, nullptr);
    f32_to_bf16_pad<<<gs(MC * 512), 256, 0, stream>>>(m1, m1bf, (int)MC, 512, 512);
    bilstm(m1bf, (int)MC, LC_, m2WihB, m2WhhB, mod2_b, 512, m2, 512, nullptr);

    // ---- output layer: concat [g, m2] -> out BiLSTM, keep only final h,c ----
    gm_build_kernel<<<gs(MC * 2560), 256, 0, stream>>>(gbf, m2, gmbf, (int)(MC * 2560));
    bilstm(gmbf, (int)MC, LC_, oWihB, oWhhB, out_b, 2560, c2q /*scratch dump*/, 512, feat);

    // ---- classifier ----
    clf_kernel<<<dim3(B_), 256, 0, stream>>>(feat, clf_w, clf_b, out);
}
